// MultiHeadAttention_8426725835235
// MI455X (gfx1250) — compile-verified, matrix operand order checked by reading, not run on hardware
//
#include <hip/hip_runtime.h>
#include <hip/hip_bf16.h>
#include <math.h>
#include <stdint.h>

// ---------------- problem constants (from reference) ----------------
#define DMODEL 1024
#define NHEADS 16
#define DK     64
#define SEQ    2048
#define BATCH  2
#define NTOK   (BATCH * SEQ)        // 4096
#define KSLAB  128                  // K-depth staged per TDM transfer

typedef __bf16 bf16_t;
typedef __attribute__((ext_vector_type(16))) __bf16 v16bf;
typedef __attribute__((ext_vector_type(8)))  __bf16 bf16x8;
typedef __attribute__((ext_vector_type(8)))  float  v8f;
typedef uint32_t u32;
typedef u32 u32x4 __attribute__((ext_vector_type(4)));
typedef u32 u32x8 __attribute__((ext_vector_type(8)));

#define WMMA_BF16(a, b, c) \
    __builtin_amdgcn_wmma_f32_16x16x32_bf16(false, (a), false, (b), (short)0, (c), false, false)

// --------------------------------------------------------------------
// 16x32 bf16 A-fragment loader (B mirrored with lane = N).
// CDNA5 16-bit A layout (ISA 7.12.2): lane l holds row M=l%16; elements
// 0..7 hold K = (l/16)*8 + 0..7, elements 8..15 hold K = 16 + (l/16)*8 + 0..7.
// Row-major storage -> two b128 loads (global_load_b128 or ds_load_b128).
// --------------------------------------------------------------------
__device__ __forceinline__ v16bf load_frag(const bf16_t* __restrict__ src,
                                           int ld, int row0, int k0) {
    int l  = threadIdx.x & 31;
    int r  = row0 + (l & 15);
    int kb = k0 + ((l >> 4) << 3);
    const bf16_t* p = src + (size_t)r * ld + kb;
    union { v16bf v; bf16x8 h[2]; } u;
    u.h[0] = *(const bf16x8*)(p);
    u.h[1] = *(const bf16x8*)(p + 16);
    return u.v;
}

// --------------------------------------------------------------------
// Tensor Data Mover: DMA a 64-row x 256-byte (KSLAB bf16) B panel into LDS.
// D# per CDNA5 ISA: 2D tensor -> groups 0,1 only
// (VADDR2/VADDR3 omitted = NULL). data_size = 4 bytes.
//   tensor_dim0 = 512 dwords (2048B row), tile_dim0 = 64 dwords (256B)
//   tensor_dim1 = 1024 rows,              tile_dim1 = 64 rows
//   tensor_dim0_stride = 512 dwords
// --------------------------------------------------------------------
__device__ __forceinline__ void tdm_load_b_panel(const bf16_t* Bt, int n0, int k,
                                                 bf16_t* ldsbuf) {
    uint64_t ga = (uint64_t)(uintptr_t)Bt + (((uint64_t)n0 * DMODEL + (uint64_t)k) << 1);
    u32x4 g0;
    g0[0] = 1u;                                   // count=1 (valid descriptor)
    g0[1] = (u32)(uintptr_t)ldsbuf;               // LDS byte addr = flat addr[31:0]
    g0[2] = (u32)ga;                              // global_addr[31:0]
    g0[3] = (u32)(ga >> 32) | (2u << 30);         // global_addr[56:32] | type=2
    u32x8 g1;
    g1[0] = (2u << 16);                           // workgroup_mask=0, data_size=4B
    g1[1] = (512u << 16);                         // tensor_dim0 (low16 @ bit48)
    g1[2] = (1024u << 16);                        // tensor_dim1 (low16 @ bit80)
    g1[3] = ((KSLAB / 2) << 16);                  // tile_dim0 = 64 dwords (256B)
    g1[4] = 64u;                                  // tile_dim1 = 64, tile_dim2 = 0
    g1[5] = 512u;                                 // tensor_dim0_stride low32
    g1[6] = 0u;
    g1[7] = 0u;
    asm volatile("tensor_load_to_lds %0, %1" :: "s"(g0), "s"(g1) : "memory");
}

// --------------------------------------------------------------------
// TDM-staged, register-blocked GEMM block: 8 waves x (32x64) = 256x64 per WG.
// B panel (shared by all 8 waves) is double-buffered in LDS via TDM;
// A fragments stream from global (distinct rows per wave). Fully unrolled
// so accumulators stay pinned in place (no loop-carried phi copies).
// --------------------------------------------------------------------
__device__ __forceinline__ void gemm_block_tdm(const bf16_t* __restrict__ A,
                                               const bf16_t* __restrict__ Bt,
                                               int m0, int n0,
                                               bf16_t* Bp /*[2][64*KSLAB]*/,
                                               v8f acc[8]) {
    int wid = threadIdx.x >> 5;
    int l   = threadIdx.x & 31;
    const int NSTAGE = DMODEL / KSLAB;            // 8

    if (wid == 0) tdm_load_b_panel(Bt, n0, 0, Bp);
    __builtin_amdgcn_s_wait_tensorcnt(0);         // no-op for waves with TENSORcnt==0
    __syncthreads();

#pragma unroll
    for (int s = 0; s < NSTAGE; ++s) {
        // kick off the next slab's DMA while we compute this one
        if (wid == 0 && s + 1 < NSTAGE)
            tdm_load_b_panel(Bt, n0, (s + 1) * KSLAB, Bp + ((s + 1) & 1) * (64 * KSLAB));

        const bf16_t* bp = Bp + (s & 1) * (64 * KSLAB);
#pragma unroll
        for (int kk = 0; kk < KSLAB; kk += 32) {
            int k = s * KSLAB + kk;
            if (k + 256 < DMODEL)                 // prefetch A ahead (global_prefetch_b8)
                __builtin_prefetch(A + (size_t)(m0 + l) * DMODEL + k + 256, 0, 1);
            v16bf a0 = load_frag(A, DMODEL, m0,      k);
            v16bf a1 = load_frag(A, DMODEL, m0 + 16, k);
#pragma unroll
            for (int j = 0; j < 4; ++j) {
                v16bf b = load_frag(bp, KSLAB, j * 16, kk);   // ds_load_b128 x2
                acc[j]     = WMMA_BF16(a0, b, acc[j]);
                acc[4 + j] = WMMA_BF16(a1, b, acc[4 + j]);
            }
        }
        __builtin_amdgcn_s_wait_tensorcnt(0);
        __syncthreads();
    }
}

// ---------------- fp32 -> bf16 conversions ----------------
__global__ void cvt_bf16_kernel(const float* __restrict__ in, bf16_t* __restrict__ out, int n) {
    int i = blockIdx.x * blockDim.x + threadIdx.x;
    if (i < n) out[i] = (bf16_t)in[i];
}

// W [K,N] row-major fp32 -> Wt [N,K] row-major bf16 (B loads contiguously)
__global__ void transpose_bf16_kernel(const float* __restrict__ in, bf16_t* __restrict__ out, int dim) {
    int i = blockIdx.x * blockDim.x + threadIdx.x;
    if (i < dim * dim) {
        int k = i / dim, n = i % dim;
        out[(size_t)n * dim + k] = (bf16_t)in[i];
    }
}

// ---------------- fused QKV projection ----------------
// One WG per 256x64 output block; which = 0/1/2 -> Q/K/V. The 64-wide
// N-block is exactly one head (DK==64), so h = nb.
// Q,K stored [b,h,s,dk]; V stored transposed [b,h,dk,s].
__global__ __launch_bounds__(256)
void qkv_proj_kernel(const bf16_t* __restrict__ xb,
                     const bf16_t* __restrict__ wq_t, const bf16_t* __restrict__ wk_t,
                     const bf16_t* __restrict__ wv_t,
                     const float* __restrict__ bq, const float* __restrict__ bk,
                     const float* __restrict__ bv,
                     bf16_t* __restrict__ Q, bf16_t* __restrict__ K, bf16_t* __restrict__ V) {
    __shared__ bf16_t Bp[2][64 * KSLAB];        // double-buffered B panel (32 KB)

    const int NB = DMODEL / 64;                 // 16 N-blocks (== heads)
    const int MB = NTOK / 256;                  // 16 M-blocks
    int nb    = blockIdx.x % NB;
    int rest  = blockIdx.x / NB;
    int mb    = rest % MB;
    int which = rest / MB;                      // 0,1,2

    const bf16_t* Wt  = (which == 0) ? wq_t : (which == 1) ? wk_t : wv_t;
    const float*  bia = (which == 0) ? bq   : (which == 1) ? bk   : bv;

    int wid = threadIdx.x >> 5;
    int m0  = mb * 256 + wid * 32;
    int n0  = nb * 64;

    v8f acc[8] = {};
    gemm_block_tdm(xb, Wt, m0, n0, &Bp[0][0], acc);

    int l = threadIdx.x & 31;
    int h = nb;                                  // head index
#pragma unroll
    for (int j = 0; j < 8; ++j) {
        int d = (j & 3) * 16 + (l & 15);         // 0..63 within head
        float bval = bia[h * DK + d];
#pragma unroll
        for (int i = 0; i < 8; ++i) {
            int t = m0 + ((j >> 2) << 4) + i + ((l >> 4) << 3);
            int b = t >> 11, s = t & (SEQ - 1);
            bf16_t o = (bf16_t)(acc[j][i] + bval);
            if (which == 2) {
                V[(((size_t)b * NHEADS + h) * DK + d) * SEQ + s] = o;
            } else {
                bf16_t* dst = (which == 0) ? Q : K;
                dst[(((size_t)b * NHEADS + h) * SEQ + s) * DK + d] = o;
            }
        }
    }
}

// ---------------- flash attention ----------------
// One wave per (b, h, 16-query block). Online softmax over 32-key blocks.
__global__ __launch_bounds__(32)
void attn_kernel(const bf16_t* __restrict__ Q, const bf16_t* __restrict__ K,
                 const bf16_t* __restrict__ V, bf16_t* __restrict__ attn_out) {
    __shared__ bf16_t Plds[16 * 32];            // P tile, row-major [M=16][key=32]

    int wave = blockIdx.x;                      // 0 .. B*H*(S/16)-1
    int qt   = wave & (SEQ / 16 - 1);           // query tile
    int bh   = wave >> 7;                       // b*H + h
    const bf16_t* Qh = Q + (size_t)bh * SEQ * DK;
    const bf16_t* Kh = K + (size_t)bh * SEQ * DK;
    const bf16_t* Vh = V + (size_t)bh * DK * SEQ;   // [dk][s]
    int q0 = qt * 16;
    int l  = threadIdx.x & 31;

    v16bf qa0 = load_frag(Qh, DK, q0, 0);
    v16bf qa1 = load_frag(Qh, DK, q0, 32);

    v8f o0 = {}, o1 = {}, o2 = {}, o3 = {};     // O accumulator, 16 x 64
    float m_run[8], l_run[8];
#pragma unroll
    for (int i = 0; i < 8; ++i) { m_run[i] = -INFINITY; l_run[i] = 0.0f; }
    const float scale = 0.125f;                 // 1/sqrt(64)

    for (int kb = 0; kb < SEQ; kb += 32) {
        if (kb + 32 < SEQ) {
            __builtin_prefetch(Kh + (size_t)(kb + 32 + l) * DK, 0, 1);
            __builtin_prefetch(Vh + (size_t)l        * SEQ + kb + 32, 0, 1);
            __builtin_prefetch(Vh + (size_t)(l + 32) * SEQ + kb + 32, 0, 1);
        }

        // ---- scores: two 16x16 tiles covering keys kb..kb+31 ----
        v8f c0 = {}, c1 = {};
        {
            v16bf b0 = load_frag(Kh, DK, kb, 0);
            v16bf b1 = load_frag(Kh, DK, kb, 32);
            c0 = WMMA_BF16(qa0, b0, c0);
            c0 = WMMA_BF16(qa1, b1, c0);
            v16bf b2 = load_frag(Kh, DK, kb + 16, 0);
            v16bf b3 = load_frag(Kh, DK, kb + 16, 32);
            c1 = WMMA_BF16(qa0, b2, c1);
            c1 = WMMA_BF16(qa1, b3, c1);
        }

        // ---- online softmax per row (row M = i + 8*(l/16); cols across 16 lanes) ----
#pragma unroll
        for (int i = 0; i < 8; ++i) {
            float s0 = c0[i] * scale;
            float s1 = c1[i] * scale;
            float v  = fmaxf(s0, s1);
#pragma unroll
            for (int off = 1; off < 16; off <<= 1)
                v = fmaxf(v, __shfl_xor(v, off, 32));
            float mn   = fmaxf(m_run[i], v);
            float corr = __expf(m_run[i] - mn);
            m_run[i]   = mn;
            float p0 = __expf(s0 - mn);
            float p1 = __expf(s1 - mn);
            float rs = p0 + p1;
#pragma unroll
            for (int off = 1; off < 16; off <<= 1)
                rs += __shfl_xor(rs, off, 32);
            l_run[i] = l_run[i] * corr + rs;
            o0[i] *= corr; o1[i] *= corr; o2[i] *= corr; o3[i] *= corr;

            int row = i + ((l >> 4) << 3);
            int col = (l & 15);
            Plds[row * 32 + col]      = (bf16_t)p0;
            Plds[row * 32 + 16 + col] = (bf16_t)p1;
        }
        __syncthreads();

        // ---- O += P (16x32) x V (32x64) ----
        v16bf pa  = load_frag(Plds, 32, 0, 0);
        v16bf vb0 = load_frag(Vh, SEQ, 0,  kb);
        v16bf vb1 = load_frag(Vh, SEQ, 16, kb);
        v16bf vb2 = load_frag(Vh, SEQ, 32, kb);
        v16bf vb3 = load_frag(Vh, SEQ, 48, kb);
        o0 = WMMA_BF16(pa, vb0, o0);
        o1 = WMMA_BF16(pa, vb1, o1);
        o2 = WMMA_BF16(pa, vb2, o2);
        o3 = WMMA_BF16(pa, vb3, o3);
        __syncthreads();
    }

    // ---- epilogue: O / l, scatter to attn_out [b, s, h*64 + d] (bf16) ----
    int b = bh >> 4, h = bh & (NHEADS - 1);
#pragma unroll
    for (int i = 0; i < 8; ++i) {
        int srow = q0 + i + ((l >> 4) << 3);
        size_t rowbase = ((size_t)b * SEQ + srow) * DMODEL + h * DK;
        float inv = 1.0f / l_run[i];
        int n = (l & 15);
        attn_out[rowbase + 0  + n] = (bf16_t)(o0[i] * inv);
        attn_out[rowbase + 16 + n] = (bf16_t)(o1[i] * inv);
        attn_out[rowbase + 32 + n] = (bf16_t)(o2[i] * inv);
        attn_out[rowbase + 48 + n] = (bf16_t)(o3[i] * inv);
    }
}

// ---------------- output projection (TDM-staged 256x64 blocks) ----------------
__global__ __launch_bounds__(256)
void out_proj_kernel(const bf16_t* __restrict__ attn, const bf16_t* __restrict__ wo_t,
                     const float* __restrict__ bo, float* __restrict__ out) {
    __shared__ bf16_t Bp[2][64 * KSLAB];

    const int NB = DMODEL / 64;                  // 16
    int nb = blockIdx.x % NB;
    int mb = blockIdx.x / NB;

    int wid = threadIdx.x >> 5;
    int m0  = mb * 256 + wid * 32;
    int n0  = nb * 64;

    v8f acc[8] = {};
    gemm_block_tdm(attn, wo_t, m0, n0, &Bp[0][0], acc);

    int l = threadIdx.x & 31;
#pragma unroll
    for (int j = 0; j < 8; ++j) {
        int n = n0 + (j & 3) * 16 + (l & 15);
        float bval = bo[n];
#pragma unroll
        for (int i = 0; i < 8; ++i) {
            int t = m0 + ((j >> 2) << 4) + i + ((l >> 4) << 3);
            out[(size_t)t * DMODEL + n] = acc[j][i] + bval;
        }
    }
}

// --------------------------------------------------------------------
extern "C" void kernel_launch(void* const* d_in, const int* in_sizes, int n_in,
                              void* d_out, int out_size, void* d_ws, size_t ws_size,
                              hipStream_t stream) {
    (void)in_sizes; (void)n_in; (void)out_size; (void)ws_size;

    const float* x  = (const float*)d_in[0];
    const float* Wq = (const float*)d_in[1];
    const float* bq = (const float*)d_in[2];
    const float* Wk = (const float*)d_in[3];
    const float* bk = (const float*)d_in[4];
    const float* Wv = (const float*)d_in[5];
    const float* bv = (const float*)d_in[6];
    const float* Wo = (const float*)d_in[7];
    const float* bo = (const float*)d_in[8];
    float* out = (float*)d_out;

    // workspace layout (bf16), total 48 MB
    char* ws = (char*)d_ws;
    bf16_t* xb   = (bf16_t*)(ws);                                 //  8 MB  x as bf16
    bf16_t* wq_t = (bf16_t*)(ws + (8u << 20));                    //  2 MB  Wq^T
    bf16_t* wk_t = (bf16_t*)(ws + (10u << 20));                   //  2 MB
    bf16_t* wv_t = (bf16_t*)(ws + (12u << 20));                   //  2 MB
    bf16_t* wo_t = (bf16_t*)(ws + (14u << 20));                   //  2 MB
    bf16_t* Qd   = (bf16_t*)(ws + (16u << 20));                   //  8 MB  [b,h,s,dk]
    bf16_t* Kd   = (bf16_t*)(ws + (24u << 20));                   //  8 MB  [b,h,s,dk]
    bf16_t* Vd   = (bf16_t*)(ws + (32u << 20));                   //  8 MB  [b,h,dk,s]
    bf16_t* At   = (bf16_t*)(ws + (40u << 20));                   //  8 MB  attn out [b,s,d]

    // 1) precision conversion / weight transposes
    {
        int n = NTOK * DMODEL;
        cvt_bf16_kernel<<<(n + 255) / 256, 256, 0, stream>>>(x, xb, n);
        int w = DMODEL * DMODEL;
        int g = (w + 255) / 256;
        transpose_bf16_kernel<<<g, 256, 0, stream>>>(Wq, wq_t, DMODEL);
        transpose_bf16_kernel<<<g, 256, 0, stream>>>(Wk, wk_t, DMODEL);
        transpose_bf16_kernel<<<g, 256, 0, stream>>>(Wv, wv_t, DMODEL);
        transpose_bf16_kernel<<<g, 256, 0, stream>>>(Wo, wo_t, DMODEL);
    }

    // 2) fused QKV projection: 3 * 16 * 16 = 768 workgroups of 256 threads
    qkv_proj_kernel<<<3 * (NTOK / 256) * (DMODEL / 64), 256, 0, stream>>>(
        xb, wq_t, wk_t, wv_t, bq, bk, bv, Qd, Kd, Vd);

    // 3) flash attention: B*H*(S/16) = 4096 single-wave workgroups
    attn_kernel<<<BATCH * NHEADS * (SEQ / 16), 32, 0, stream>>>(Qd, Kd, Vd, At);

    // 4) output projection -> fp32 d_out: 256 workgroups
    out_proj_kernel<<<(NTOK / 256) * (DMODEL / 64), 256, 0, stream>>>(At, wo_t, bo, out);
}
